// Encoder_62294205661429
// MI455X (gfx1250) — matrix operand order repaired and checked by reading, not compiled, not run on hardware
//
#include <hip/hip_runtime.h>

typedef __attribute__((ext_vector_type(16))) __bf16 v16bf;
typedef __attribute__((ext_vector_type(8)))  float  v8f;

#define HID       128
#define LN_EPS    1e-5f
#define NT        8            // 8 n-tiles of 16 cols = 128
#define TILE_ROWS 128          // rows per block (8 waves x 16 rows)
#define BLOCK     256
#define HSTRIDE   136          // LDS row stride for hidden (128 + 8 pad), 16B-aligned

// ---------------------------------------------------------------------------
// Weight packing: W[K x 128] fp32 row-major -> bf16 B-fragment layout.
// Fragment element index: ((kc*NT + nt)*32 + lane), 16 bf16 per lane.
// lane<16: col = nt*16+lane ; lane>=16: col = nt*16+lane-16, half = lane>>4,
// k = kc*32 + half*8 + e (e<8), kc*32 + 16 + half*8 + e-8 (e>=8).
// ---------------------------------------------------------------------------
__global__ void pack_weights_kernel(const float* __restrict__ W, int K,
                                    v16bf* __restrict__ out, int Kp) {
  int idx = blockIdx.x * blockDim.x + threadIdx.x;
  int total = (Kp >> 5) * NT * 32;
  if (idx >= total) return;
  int lane = idx & 31;
  int nt   = (idx >> 5) & 7;
  int kc   = idx >> 8;
  int col   = nt * 16 + (lane & 15);
  int kbase = kc * 32 + (lane >> 4) * 8;
  v16bf v;
#pragma unroll
  for (int e = 0; e < 16; ++e) {
    int k = kbase + (e & 7) + ((e >> 3) << 4);
    float f = (k < K) ? W[(size_t)k * HID + col] : 0.f;
    v[e] = (__bf16)f;
  }
  out[idx] = v;
}

__global__ void zero_f32(float* __restrict__ p, long long n) {
  long long i  = (long long)blockIdx.x * blockDim.x + threadIdx.x;
  long long st = (long long)gridDim.x * blockDim.x;
  for (; i < n; i += st) p[i] = 0.f;
}

// ---------------------------------------------------------------------------
// Gathers: provide 8 consecutive fp32 input features starting at k (k % 8 == 0).
// All concat boundaries are multiples of 8, so a run never straddles segments.
// ---------------------------------------------------------------------------
struct GEdgeFeat {          // [dist0, dist1, norm] , K=3 padded to 32
  float d0, d1, nrm;
  __device__ void load8(int k, float* o) const {
#pragma unroll
    for (int j = 0; j < 8; ++j) {
      int kk = k + j;
      o[j] = (kk == 0) ? d0 : (kk == 1) ? d1 : (kk == 2) ? nrm : 0.f;
    }
  }
};

struct GNodeFeat {          // [states(3), node_type(9)] , K=12 padded to 32
  const float *st, *nt;
  __device__ void load8(int k, float* o) const {
#pragma unroll
    for (int j = 0; j < 8; ++j) {
      int kk = k + j;
      o[j] = (kk < 3) ? st[kk] : (kk < 12) ? nt[kk - 3] : 0.f;
    }
  }
};

struct GConcatEdge {        // [V|pos (s) 184, V|pos (r) 184, E 128] K=496 pad 512
  const float *vs, *ps, *vr, *pr, *er;
  __device__ void load8(int k, float* o) const {
    const float* src;
    if (k < 368) {
      int kk = k; const float* v = vs; const float* p = ps;
      if (k >= 184) { kk = k - 184; v = vr; p = pr; }
      src = (kk < 128) ? (v + kk) : (p + (kk - 128));
    } else if (k < 496) {
      src = er + (k - 368);
    } else {
#pragma unroll
      for (int j = 0; j < 8; ++j) o[j] = 0.f;
      return;
    }
    float4 a = *(const float4*)(src);
    float4 b = *(const float4*)(src + 4);
    o[0]=a.x; o[1]=a.y; o[2]=a.z; o[3]=a.w; o[4]=b.x; o[5]=b.y; o[6]=b.z; o[7]=b.w;
  }
};

struct GConcatNode {        // [V 128, pos 56, agg 128] K=312 pad 320
  const float *v, *p, *a;
  __device__ void load8(int k, float* o) const {
    const float* src;
    if (k < 128)      src = v + k;
    else if (k < 184) src = p + (k - 128);
    else if (k < 312) src = a + (k - 184);
    else {
#pragma unroll
      for (int j = 0; j < 8; ++j) o[j] = 0.f;
      return;
    }
    float4 x = *(const float4*)(src);
    float4 y = *(const float4*)(src + 4);
    o[0]=x.x; o[1]=x.y; o[2]=x.z; o[3]=x.w; o[4]=y.x; o[5]=y.y; o[6]=y.z; o[7]=y.w;
  }
};

// ---------------------------------------------------------------------------
// Core fused MLP: y = [LN]( relu(X@W0+b0) @ W1 + b1 ), per-wave 16x128 tile.
// ---------------------------------------------------------------------------
template <typename G, bool HAS_LN>
__device__ __forceinline__ void mlp_forward(
    const G& g, int Kp,
    const v16bf* __restrict__ w0p, const float* __restrict__ b0,
    const v16bf* __restrict__ w1p, const float* __restrict__ b1,
    const float* __restrict__ ln_s, const float* __restrict__ ln_b,
    __bf16* hlds, float (&y)[NT][8]) {
  const int lane = threadIdx.x & 31;
  const int wid  = threadIdx.x >> 5;
  const int hlf  = lane >> 4;
  const int cl   = lane & 15;

  v8f acc[NT] = {};

  // GEMM1: gathered A (16 x Kp) x packed W0 (Kp x 128)
  for (int kc = 0; kc < (Kp >> 5); ++kc) {
    float t[16];
    g.load8(kc * 32 + hlf * 8, t);          // A layout: e<8  -> k0 + half*8 + e
    g.load8(kc * 32 + 16 + hlf * 8, t + 8); //           e>=8 -> k0+16+half*8 + e-8
    v16bf a;
#pragma unroll
    for (int e = 0; e < 16; ++e) a[e] = (__bf16)t[e];
    const v16bf* wb = w0p + (size_t)kc * NT * 32 + lane;
#pragma unroll
    for (int n = 0; n < NT; ++n) {
      v16bf b = wb[n * 32];
      acc[n] = __builtin_amdgcn_wmma_f32_16x16x32_bf16(false, a, false, b,
                                                       (short)0, acc[n], false, false);
    }
  }

  // Hidden: relu(acc + b0) -> LDS (bf16), re-layout C -> A
#pragma unroll
  for (int n = 0; n < NT; ++n) {
    float bb = b0[n * 16 + cl];
#pragma unroll
    for (int i = 0; i < 8; ++i) {
      float h = fmaxf(acc[n][i] + bb, 0.f);
      hlds[(size_t)(wid * 16 + hlf * 8 + i) * HSTRIDE + n * 16 + cl] = (__bf16)h;
    }
  }
  __syncthreads();

  v8f z = {};
#pragma unroll
  for (int n = 0; n < NT; ++n) acc[n] = z;

  // GEMM2: H (16 x 128) x packed W1 (128 x 128)
  const __bf16* hrow = hlds + (size_t)(wid * 16 + cl) * HSTRIDE;
#pragma unroll
  for (int kc = 0; kc < (HID >> 5); ++kc) {
    union { uint4 u[2]; v16bf v; } cv;
    cv.u[0] = *(const uint4*)(hrow + kc * 32 + hlf * 8);
    cv.u[1] = *(const uint4*)(hrow + kc * 32 + 16 + hlf * 8);
    v16bf a = cv.v;
    const v16bf* wb = w1p + (size_t)kc * NT * 32 + lane;
#pragma unroll
    for (int n = 0; n < NT; ++n) {
      v16bf b = wb[n * 32];
      acc[n] = __builtin_amdgcn_wmma_f32_16x16x32_bf16(false, a, false, b,
                                                       (short)0, acc[n], false, false);
    }
  }

#pragma unroll
  for (int n = 0; n < NT; ++n) {
    float bb = b1[n * 16 + cl];
#pragma unroll
    for (int i = 0; i < 8; ++i) y[n][i] = acc[n][i] + bb;
  }

  if constexpr (HAS_LN) {
    // row i lives in 16 lanes of this half-wave (8 cols each) -> width-16 reduce
#pragma unroll
    for (int i = 0; i < 8; ++i) {
      float s1 = 0.f, s2 = 0.f;
#pragma unroll
      for (int n = 0; n < NT; ++n) { float v = y[n][i]; s1 += v; s2 += v * v; }
#pragma unroll
      for (int m = 1; m < 16; m <<= 1) {
        s1 += __shfl_xor(s1, m, 16);
        s2 += __shfl_xor(s2, m, 16);
      }
      float mu  = s1 * (1.f / 128.f);
      float var = s2 * (1.f / 128.f) - mu * mu;
      float rs  = rsqrtf(var + LN_EPS);
#pragma unroll
      for (int n = 0; n < NT; ++n) {
        int col = n * 16 + cl;
        y[n][i] = (y[n][i] - mu) * rs * ln_s[col] + ln_b[col];
      }
    }
  }
}

// ---------------------------------------------------------------------------
// Kernels
// ---------------------------------------------------------------------------
__global__ __launch_bounds__(BLOCK) void enc_node_kernel(
    const float* __restrict__ states, const float* __restrict__ node_type,
    const v16bf* __restrict__ w0p, const float* __restrict__ b0,
    const v16bf* __restrict__ w1p, const float* __restrict__ b1,
    float* __restrict__ V, int N) {
  __shared__ __bf16 hlds[TILE_ROWS * HSTRIDE];
  const int tiles = (N + TILE_ROWS - 1) / TILE_ROWS;
  const int b = blockIdx.x / tiles;
  const int row0 = (blockIdx.x % tiles) * TILE_ROWS;
  const int lane = threadIdx.x & 31, wid = threadIdx.x >> 5;
  const int hlf = lane >> 4, cl = lane & 15;

  int arow = row0 + wid * 16 + cl;
  if (arow >= N) arow = N - 1;
  GNodeFeat g{states + ((long long)b * N + arow) * 3,
              node_type + ((long long)b * N + arow) * 9};
  float y[NT][8];
  mlp_forward<GNodeFeat, false>(g, 32, w0p, b0, w1p, b1, nullptr, nullptr, hlds, y);

  float* Vb = V + (long long)b * N * HID;
#pragma unroll
  for (int i = 0; i < 8; ++i) {
    int orow = row0 + wid * 16 + hlf * 8 + i;
    if (orow < N) {
#pragma unroll
      for (int n = 0; n < NT; ++n)
        Vb[(long long)orow * HID + n * 16 + cl] = y[n][i];
    }
  }
}

__global__ __launch_bounds__(BLOCK) void enc_edge_kernel(
    const float* __restrict__ mesh_pos, const long long* __restrict__ edges,
    const v16bf* __restrict__ w0p, const float* __restrict__ b0,
    const v16bf* __restrict__ w1p, const float* __restrict__ b1,
    float* __restrict__ E, int N, int Ne) {
  __shared__ __bf16 hlds[TILE_ROWS * HSTRIDE];
  const int tiles = (Ne + TILE_ROWS - 1) / TILE_ROWS;
  const int b = blockIdx.x / tiles;
  const int row0 = (blockIdx.x % tiles) * TILE_ROWS;
  const int lane = threadIdx.x & 31, wid = threadIdx.x >> 5;
  const int hlf = lane >> 4, cl = lane & 15;

  int arow = row0 + wid * 16 + cl;
  if (arow >= Ne) arow = Ne - 1;
  long long ei = (long long)b * Ne + arow;
  long long s = edges[ei * 2 + 0], r = edges[ei * 2 + 1];
  const float* ps = mesh_pos + ((long long)b * N + s) * 2;
  const float* pr = mesh_pos + ((long long)b * N + r) * 2;
  GEdgeFeat g;
  g.d0 = ps[0] - pr[0];
  g.d1 = ps[1] - pr[1];
  g.nrm = sqrtf(g.d0 * g.d0 + g.d1 * g.d1);

  float y[NT][8];
  mlp_forward<GEdgeFeat, false>(g, 32, w0p, b0, w1p, b1, nullptr, nullptr, hlds, y);

  float* Eb = E + (long long)b * Ne * HID;
#pragma unroll
  for (int i = 0; i < 8; ++i) {
    int orow = row0 + wid * 16 + hlf * 8 + i;
    if (orow < Ne) {
#pragma unroll
      for (int n = 0; n < NT; ++n)
        Eb[(long long)orow * HID + n * 16 + cl] = y[n][i];
    }
  }
}

__global__ __launch_bounds__(BLOCK) void f_edge_kernel(
    const float* __restrict__ V, const float* __restrict__ pos,
    float* __restrict__ E, const long long* __restrict__ edges,
    const v16bf* __restrict__ w0p, const float* __restrict__ b0,
    const v16bf* __restrict__ w1p, const float* __restrict__ b1,
    const float* __restrict__ ln_s, const float* __restrict__ ln_b,
    float* __restrict__ agg, int N, int Ne) {
  __shared__ __bf16 hlds[TILE_ROWS * HSTRIDE];
  const int tiles = (Ne + TILE_ROWS - 1) / TILE_ROWS;
  const int b = blockIdx.x / tiles;
  const int row0 = (blockIdx.x % tiles) * TILE_ROWS;
  const int lane = threadIdx.x & 31, wid = threadIdx.x >> 5;
  const int hlf = lane >> 4, cl = lane & 15;

  const float* Vb = V + (long long)b * N * HID;
  const float* Pb = pos + (long long)b * N * 56;
  float* Eb = E + (long long)b * Ne * HID;
  float* Ab = agg + (long long)b * N * HID;

  int arow = row0 + wid * 16 + cl;
  if (arow >= Ne) arow = Ne - 1;
  long long ei = (long long)b * Ne + arow;
  long long s = edges[ei * 2 + 0], r = edges[ei * 2 + 1];
  GConcatEdge g{Vb + s * HID, Pb + s * 56, Vb + r * HID, Pb + r * 56,
                Eb + (long long)arow * HID};

  float y[NT][8];
  mlp_forward<GConcatEdge, true>(g, 512, w0p, b0, w1p, b1, ln_s, ln_b, hlds, y);

  // epilogue: E += e_new (in place), agg[r] += e_new (scatter)
#pragma unroll
  for (int i = 0; i < 8; ++i) {
    int orow = row0 + wid * 16 + hlf * 8 + i;
    int oc = (orow < Ne) ? orow : (Ne - 1);
    long long rr = edges[((long long)b * Ne + oc) * 2 + 1];
    bool ok = orow < Ne;
#pragma unroll
    for (int n = 0; n < NT; ++n) {
      if (ok) {
        int col = n * 16 + cl;
        float en = y[n][i];
        long long eo = (long long)orow * HID + col;
        Eb[eo] = Eb[eo] + en;
        __hip_atomic_fetch_add(&Ab[rr * HID + col], en,
                               __ATOMIC_RELAXED, __HIP_MEMORY_SCOPE_AGENT);
      }
    }
  }
}

__global__ __launch_bounds__(BLOCK) void f_node_kernel(
    float* __restrict__ V, const float* __restrict__ pos,
    const float* __restrict__ agg,
    const v16bf* __restrict__ w0p, const float* __restrict__ b0,
    const v16bf* __restrict__ w1p, const float* __restrict__ b1,
    const float* __restrict__ ln_s, const float* __restrict__ ln_b,
    int N) {
  __shared__ __bf16 hlds[TILE_ROWS * HSTRIDE];
  const int tiles = (N + TILE_ROWS - 1) / TILE_ROWS;
  const int b = blockIdx.x / tiles;
  const int row0 = (blockIdx.x % tiles) * TILE_ROWS;
  const int lane = threadIdx.x & 31, wid = threadIdx.x >> 5;
  const int hlf = lane >> 4, cl = lane & 15;

  float* Vb = V + (long long)b * N * HID;
  const float* Pb = pos + (long long)b * N * 56;
  const float* Ab = agg + (long long)b * N * HID;

  int arow = row0 + wid * 16 + cl;
  if (arow >= N) arow = N - 1;
  GConcatNode g{Vb + (long long)arow * HID, Pb + (long long)arow * 56,
                Ab + (long long)arow * HID};

  float y[NT][8];
  mlp_forward<GConcatNode, true>(g, 320, w0p, b0, w1p, b1, ln_s, ln_b, hlds, y);

#pragma unroll
  for (int i = 0; i < 8; ++i) {
    int orow = row0 + wid * 16 + hlf * 8 + i;
    if (orow < N) {
#pragma unroll
      for (int n = 0; n < NT; ++n) {
        long long vo = (long long)orow * HID + n * 16 + cl;
        Vb[vo] = Vb[vo] + y[n][i];
      }
    }
  }
}

// ---------------------------------------------------------------------------
// Launch
// ---------------------------------------------------------------------------
extern "C" void kernel_launch(void* const* d_in, const int* in_sizes, int n_in,
                              void* d_out, int out_size, void* d_ws, size_t ws_size,
                              hipStream_t stream) {
  const int B  = 2;
  const int N  = in_sizes[0] / (B * 2);   // mesh_pos [B,N,2]
  const int Ne = in_sizes[1] / (B * 2);   // edges    [B,Ne,2]

  const float*     mesh_pos  = (const float*)d_in[0];
  const long long* edges     = (const long long*)d_in[1];
  const float*     states    = (const float*)d_in[2];
  const float*     node_type = (const float*)d_in[3];
  const float*     pos_enc   = (const float*)d_in[4];

  // params flattened pytree-style (sorted dict keys):
  // enc_edge{b0,b1,w0,w1}, enc_node{b0,b1,w0,w1},
  // gn[i]: f_edge{b0,b1,ln_b,ln_s,w0,w1}, f_node{b0,b1,ln_b,ln_s,w0,w1}
  const float* ee_b0 = (const float*)d_in[5];
  const float* ee_b1 = (const float*)d_in[6];
  const float* ee_w0 = (const float*)d_in[7];
  const float* ee_w1 = (const float*)d_in[8];
  const float* en_b0 = (const float*)d_in[9];
  const float* en_b1 = (const float*)d_in[10];
  const float* en_w0 = (const float*)d_in[11];
  const float* en_w1 = (const float*)d_in[12];

  float* V = (float*)d_out;                       // [B,N,128]
  float* E = V + (size_t)B * N * HID;             // [B,Ne,128]

  float* agg = (float*)d_ws;                      // [B,N,128]
  size_t off = (((size_t)B * N * HID * sizeof(float)) + 255) & ~(size_t)255;

  auto packW = [&](const float* W, int K, int Kp) -> const v16bf* {
    v16bf* dst = (v16bf*)((char*)d_ws + off);
    off += (size_t)(Kp >> 5) * NT * 32 * sizeof(v16bf);
    int total = (Kp >> 5) * NT * 32;
    pack_weights_kernel<<<(total + 255) / 256, 256, 0, stream>>>(W, K, dst, Kp);
    return dst;
  };

  const v16bf* ee_w0p = packW(ee_w0, 3, 32);
  const v16bf* ee_w1p = packW(ee_w1, 128, 128);
  const v16bf* en_w0p = packW(en_w0, 12, 32);
  const v16bf* en_w1p = packW(en_w1, 128, 128);

  const float *feb0[4], *feb1[4], *felnb[4], *felns[4];
  const float *fnb0[4], *fnb1[4], *fnlnb[4], *fnlns[4];
  const v16bf *few0p[4], *few1p[4], *fnw0p[4], *fnw1p[4];
  for (int l = 0; l < 4; ++l) {
    int base = 13 + l * 12;
    feb0[l]  = (const float*)d_in[base + 0];
    feb1[l]  = (const float*)d_in[base + 1];
    felnb[l] = (const float*)d_in[base + 2];
    felns[l] = (const float*)d_in[base + 3];
    few0p[l] = packW((const float*)d_in[base + 4], 496, 512);
    few1p[l] = packW((const float*)d_in[base + 5], 128, 128);
    fnb0[l]  = (const float*)d_in[base + 6];
    fnb1[l]  = (const float*)d_in[base + 7];
    fnlnb[l] = (const float*)d_in[base + 8];
    fnlns[l] = (const float*)d_in[base + 9];
    fnw0p[l] = packW((const float*)d_in[base + 10], 312, 320);
    fnw1p[l] = packW((const float*)d_in[base + 11], 128, 128);
  }

  const int tiles_n = (N + TILE_ROWS - 1) / TILE_ROWS;
  const int tiles_e = (Ne + TILE_ROWS - 1) / TILE_ROWS;

  enc_node_kernel<<<B * tiles_n, BLOCK, 0, stream>>>(
      states, node_type, en_w0p, en_b0, en_w1p, en_b1, V, N);
  enc_edge_kernel<<<B * tiles_e, BLOCK, 0, stream>>>(
      mesh_pos, edges, ee_w0p, ee_b0, ee_w1p, ee_b1, E, N, Ne);

  long long aggN = (long long)B * N * HID;
  for (int l = 0; l < 4; ++l) {
    zero_f32<<<2048, 256, 0, stream>>>(agg, aggN);
    f_edge_kernel<<<B * tiles_e, BLOCK, 0, stream>>>(
        V, pos_enc, E, edges, few0p[l], feb0[l], few1p[l], feb1[l],
        felns[l], felnb[l], agg, N, Ne);
    f_node_kernel<<<B * tiles_n, BLOCK, 0, stream>>>(
        V, pos_enc, agg, fnw0p[l], fnb0[l], fnw1p[l], fnb1[l],
        fnlns[l], fnlnb[l], N);
  }
}
